// GrapherModule_v2_85615878078952
// MI455X (gfx1250) — matrix operand order repaired and checked by reading, not compiled, not run on hardware
//
#include <hip/hip_runtime.h>
#include <hip/hip_bf16.h>
#include <math.h>

typedef float  v2f     __attribute__((ext_vector_type(2)));
typedef float  v8f     __attribute__((ext_vector_type(8)));
typedef __bf16 bf16x8  __attribute__((ext_vector_type(8)));
typedef __bf16 bf16x16 __attribute__((ext_vector_type(16)));

#define BATCH 2
#define NPTS  8192
#define CH    128
#define MROWS (BATCH * NPTS)
#define KNN   9
#define FLT_BIG 3.4e38f

// ---------------------------------------------------------------------------
// f32 WMMA GEMM: out[M,N] = A[M,K] @ W[N,K]^T + bias.  One wave computes a
// 64(M) x 16(N) tile: 4 accumulators share one B fragment per k-step.
// Software-pipelined: next k-step's fragments are fetched (branchless wrap on
// the final step) before the current step's 4 WMMAs are issued, so the
// s_wait before each WMMA covers loads issued one iteration earlier.
// ---------------------------------------------------------------------------
__global__ __launch_bounds__(256) void gemm_nt_wmma(
    const float* __restrict__ A, const float* __restrict__ W,
    const float* __restrict__ bias, float* __restrict__ out,
    int M, int N, int K, int wavesM) {
  int tid  = threadIdx.x;
  int wave = blockIdx.x * 8 + (tid >> 5);
  int lane = tid & 31;
  int half = lane >> 4;
  int r16  = lane & 15;
  int mt = wave % wavesM;
  int nt = wave / wavesM;
  int m0 = mt * 64;
  int n0 = nt * 16;

  v8f acc[4];
#pragma unroll
  for (int t = 0; t < 4; ++t) acc[t] = (v8f){0.f,0.f,0.f,0.f,0.f,0.f,0.f,0.f};

  const float* Arow = A + (size_t)(m0 + r16) * K;
  const float* Wrow = W + (size_t)(n0 + r16) * K;
  const int KSTEPS = K >> 2;

  // Prologue: fragments for k-step 0.
  v2f bfc = *(const v2f*)(Wrow + 2 * half);
  v2f afc[4];
#pragma unroll
  for (int t = 0; t < 4; ++t)
    afc[t] = *(const v2f*)(Arow + (size_t)t * 16 * K + 2 * half);

  for (int ks = 0; ks < KSTEPS; ++ks) {
    int kn = ((ks + 1 < KSTEPS) ? (ks + 1) * 4 : 0) + 2 * half;  // wrap: dummy
    v2f bfn = *(const v2f*)(Wrow + kn);
    v2f afn[4];
#pragma unroll
    for (int t = 0; t < 4; ++t)
      afn[t] = *(const v2f*)(Arow + (size_t)t * 16 * K + kn);

#pragma unroll
    for (int t = 0; t < 4; ++t)
      acc[t] = __builtin_amdgcn_wmma_f32_16x16x4_f32(
          false, afc[t], false, bfc, (short)0, acc[t], false, false);

    bfc = bfn;
#pragma unroll
    for (int t = 0; t < 4; ++t) afc[t] = afn[t];
  }

  float bj = bias[n0 + r16];
#pragma unroll
  for (int t = 0; t < 4; ++t) {
#pragma unroll
    for (int v = 0; v < 8; ++v) {
      int row = m0 + t * 16 + 8 * half + v;
      out[(size_t)row * N + n0 + r16] = acc[t][v] + bj;
    }
  }
}

// ---------------------------------------------------------------------------
// BatchNorm batch statistics: one block per feature column.
// ---------------------------------------------------------------------------
__global__ __launch_bounds__(256) void bn_stats(
    const float* __restrict__ buf, float* __restrict__ mean,
    float* __restrict__ rstd) {
  __shared__ float ss[256], ss2[256];
  int f = blockIdx.x;
  float s = 0.f, s2 = 0.f;
  for (int r = threadIdx.x; r < MROWS; r += 256) {
    float v = buf[(size_t)r * CH + f];
    s += v; s2 += v * v;
  }
  ss[threadIdx.x] = s; ss2[threadIdx.x] = s2;
  __syncthreads();
  for (int st = 128; st > 0; st >>= 1) {
    if (threadIdx.x < st) {
      ss[threadIdx.x]  += ss[threadIdx.x + st];
      ss2[threadIdx.x] += ss2[threadIdx.x + st];
    }
    __syncthreads();
  }
  if (threadIdx.x == 0) {
    float m   = ss[0] / (float)MROWS;
    float var = ss2[0] / (float)MROWS - m * m;
    mean[f] = m;
    rstd[f] = rsqrtf(var + 1e-5f);
  }
}

// BN apply (fc1): write f32 h and bf16 h for the KNN WMMA.
__global__ __launch_bounds__(256) void bn_apply_h(
    const float* __restrict__ pre, const float* __restrict__ mean,
    const float* __restrict__ rstd, const float* __restrict__ g,
    const float* __restrict__ b, float* __restrict__ h,
    __bf16* __restrict__ hb) {
  size_t i = (size_t)blockIdx.x * 256 + threadIdx.x;
  int f = (int)(i & (CH - 1));
  float v = (pre[i] - mean[f]) * rstd[f] * g[f] + b[f];
  h[i]  = v;
  hb[i] = (__bf16)v;
}

// Per-row squared norms (one wave per row).
__global__ __launch_bounds__(256) void rowsq(
    const float* __restrict__ h, float* __restrict__ xsq) {
  int row  = blockIdx.x * 8 + (threadIdx.x >> 5);
  int lane = threadIdx.x & 31;
  const float* p = h + (size_t)row * CH;
  float s = 0.f;
  for (int c = lane; c < CH; c += 32) { float v = p[c]; s += v * v; }
  for (int off = 16; off > 0; off >>= 1) s += __shfl_xor(s, off, 32);
  if (lane == 0) xsq[row] = s;
}

// ---------------------------------------------------------------------------
// Fused KNN: one block (4 waves) per 16 queries of one batch. Each wave
// streams point tiles, v_wmma_f32_16x16x32_bf16 accumulates q.x dot products;
// D layout puts one query column per lane -> per-lane sorted top-9 registers,
// merged across the 8 lane-groups per query through LDS.
// Software-pipelined: next point-tile's A fragments + squared norms are
// fetched before the current tile's WMMAs and top-9 insertion, so the
// insertion VALU work overlaps the in-flight loads. The prefetch wrap is
// wave-uniform, keeping EXEC all-ones around the WMMAs.
// ---------------------------------------------------------------------------
__global__ __launch_bounds__(128) void knn_kernel(
    const __bf16* __restrict__ hb, const float* __restrict__ xsq,
    int* __restrict__ eidx) {
  __shared__ float s_d[16 * 72];
  __shared__ int   s_i[16 * 72];

  int tid  = threadIdx.x;
  int wv   = tid >> 5;
  int lane = tid & 31;
  int half = lane >> 4;
  int r16  = lane & 15;

  int blk   = blockIdx.x;
  int batch = blk >> 9;          // 512 query tiles per batch
  int qt    = blk & 511;
  int q0    = qt * 16;

  const __bf16* hbase = hb + (size_t)batch * NPTS * CH;
  const float*  xs    = xsq + batch * NPTS;

  // B fragments: queries, 4 K-windows of 32. Lane n=r16 holds K=k0+8*half..+7 and k0+16+8*half..+7.
  bf16x16 Bf[4];
  const __bf16* qrow = hbase + (size_t)(q0 + r16) * CH + half * 8;
#pragma unroll
  for (int kw = 0; kw < 4; ++kw) {
    union { bf16x16 v; bf16x8 h[2]; } ub;
    ub.h[0] = *(const bf16x8*)(qrow + kw * 32);
    ub.h[1] = *(const bf16x8*)(qrow + kw * 32 + 16);
    Bf[kw] = ub.v;
  }

  float qs = xs[q0 + r16];
  float bd[KNN]; int bi[KNN];
#pragma unroll
  for (int j = 0; j < KNN; ++j) { bd[j] = FLT_BIG; bi[j] = 0; }

  // Prologue: fragments + norms for the wave's first point tile.
  bf16x16 afc[4];
  float   xsc[8];
  {
    const __bf16* arow = hbase + (size_t)(wv * 16 + r16) * CH + half * 8;
#pragma unroll
    for (int kw = 0; kw < 4; ++kw) {
      union { bf16x16 v; bf16x8 h[2]; } u;
      u.h[0] = *(const bf16x8*)(arow + kw * 32);
      u.h[1] = *(const bf16x8*)(arow + kw * 32 + 16);
      afc[kw] = u.v;
    }
    int pb = wv * 16 + 8 * half;
#pragma unroll
    for (int v = 0; v < 8; ++v) xsc[v] = xs[pb + v];
  }

  for (int pt = wv; pt < NPTS / 16; pt += 4) {   // wave-uniform trip count
    // Prefetch next tile (wrap to first tile on last iteration: dummy loads).
    int ptn = (pt + 4 < NPTS / 16) ? (pt + 4) : wv;
    int p0n = ptn * 16;
    bf16x16 afn[4];
    float   xsn[8];
    {
      const __bf16* arow = hbase + (size_t)(p0n + r16) * CH + half * 8;
#pragma unroll
      for (int kw = 0; kw < 4; ++kw) {
        union { bf16x16 v; bf16x8 h[2]; } u;
        u.h[0] = *(const bf16x8*)(arow + kw * 32);
        u.h[1] = *(const bf16x8*)(arow + kw * 32 + 16);
        afn[kw] = u.v;
      }
      int pbn = p0n + 8 * half;
#pragma unroll
      for (int v = 0; v < 8; ++v) xsn[v] = xs[pbn + v];
    }

    v8f acc = (v8f){0.f,0.f,0.f,0.f,0.f,0.f,0.f,0.f};
#pragma unroll
    for (int kw = 0; kw < 4; ++kw)
      acc = __builtin_amdgcn_wmma_f32_16x16x32_bf16(
          false, afc[kw], false, Bf[kw], (short)0, acc, false, false);

    int pb = pt * 16 + 8 * half;
#pragma unroll
    for (int v = 0; v < 8; ++v) {
      int   pidx = pb + v;
      float d = qs + xsc[v] - 2.0f * acc[v];
      if (d < bd[KNN - 1]) {            // sorted-register insertion
        float cd = d; int ci = pidx;
#pragma unroll
        for (int j = 0; j < KNN; ++j) {
          bool  lt = cd < bd[j];
          float td = bd[j]; int ti = bi[j];
          bd[j] = lt ? cd : td;  bi[j] = lt ? ci : ti;
          cd    = lt ? td : cd;  ci    = lt ? ti : ci;
        }
      }
    }

#pragma unroll
    for (int kw = 0; kw < 4; ++kw) afc[kw] = afn[kw];
#pragma unroll
    for (int v = 0; v < 8; ++v) xsc[v] = xsn[v];
  }

  int base = r16 * 72 + (wv * 2 + half) * 9;
#pragma unroll
  for (int j = 0; j < KNN; ++j) { s_d[base + j] = bd[j]; s_i[base + j] = bi[j]; }
  __syncthreads();

  if (tid < 16) {
    float* pd = s_d + tid * 72;
    int*   pi = s_i + tid * 72;
    int*   eo = eidx + ((size_t)batch * NPTS + q0 + tid) * KNN;
    for (int kk = 0; kk < KNN; ++kk) {
      float m = pd[0]; int mp = 0;
      for (int t = 1; t < 72; ++t)
        if (pd[t] < m) { m = pd[t]; mp = t; }
      eo[kk] = pi[mp];
      pd[mp] = FLT_BIG;
    }
  }
}

// Max-relative aggregation + concat: cat[r] = [h, max_j h[nbr_j] - h].
__global__ __launch_bounds__(128) void maxrel(
    const float* __restrict__ h, const int* __restrict__ eidx,
    float* __restrict__ cat) {
  int r     = blockIdx.x;
  int batch = r >> 13;           // NPTS = 8192
  int c     = threadIdx.x;
  const float* hbase = h + (size_t)batch * NPTS * CH;
  float hv = h[(size_t)r * CH + c];
  float m  = -FLT_BIG;
#pragma unroll
  for (int kk = 0; kk < KNN; ++kk) {
    int id = eidx[(size_t)r * KNN + kk];
    m = fmaxf(m, hbase[(size_t)id * CH + c]);
  }
  cat[(size_t)r * (2 * CH) + c]      = hv;
  cat[(size_t)r * (2 * CH) + CH + c] = m - hv;
}

// BN + exact GELU.
__global__ __launch_bounds__(256) void bn_gelu(
    const float* __restrict__ pre, const float* __restrict__ mean,
    const float* __restrict__ rstd, const float* __restrict__ g,
    const float* __restrict__ b, float* __restrict__ out) {
  size_t i = (size_t)blockIdx.x * 256 + threadIdx.x;
  int f = (int)(i & (CH - 1));
  float v = (pre[i] - mean[f]) * rstd[f] * g[f] + b[f];
  out[i] = 0.5f * v * (1.0f + erff(v * 0.70710678118654752f));
}

// BN + residual.
__global__ __launch_bounds__(256) void bn_res(
    const float* __restrict__ pre, const float* __restrict__ mean,
    const float* __restrict__ rstd, const float* __restrict__ g,
    const float* __restrict__ b, const float* __restrict__ x,
    float* __restrict__ out) {
  size_t i = (size_t)blockIdx.x * 256 + threadIdx.x;
  int f = (int)(i & (CH - 1));
  out[i] = (pre[i] - mean[f]) * rstd[f] * g[f] + b[f] + x[i];
}

// ---------------------------------------------------------------------------
extern "C" void kernel_launch(void* const* d_in, const int* in_sizes, int n_in,
                              void* d_out, int out_size, void* d_ws,
                              size_t ws_size, hipStream_t stream) {
  (void)in_sizes; (void)n_in; (void)out_size; (void)ws_size;

  const float* x      = (const float*)d_in[0];
  const float* fc1_w  = (const float*)d_in[1];
  const float* fc1_b  = (const float*)d_in[2];
  const float* bn1_g  = (const float*)d_in[3];
  const float* bn1_b  = (const float*)d_in[4];
  const float* conv_w = (const float*)d_in[5];
  const float* conv_b = (const float*)d_in[6];
  const float* bnc_g  = (const float*)d_in[7];
  const float* bnc_b  = (const float*)d_in[8];
  const float* fc2_w  = (const float*)d_in[9];
  const float* fc2_b  = (const float*)d_in[10];
  const float* bn2_g  = (const float*)d_in[11];
  const float* bn2_b  = (const float*)d_in[12];

  char* ws = (char*)d_ws;
  // Workspace layout (aliased stages, ~38.4 MB total):
  float*  bufA = (float*)(ws);                       // 8 MB: h_pre -> c_pre
  float*  bufB = (float*)(ws + 8388608);             // 8 MB: h -> g
  __bf16* hbf  = (__bf16*)(ws + 16777216);           // 4 MB: bf16 h for KNN
  float*  xsq  = (float*)(ws + 20971520);            // 64 KB
  int*    eidx = (int*)(ws + 21037056);              // 576 KB
  float*  cat  = (float*)(ws + 21626880);            // 16 MB; later aliased as o_pre
  float*  opre = cat;                                // fc2 pre-BN (cat is dead then)
  float*  st   = (float*)(ws + 38404096);            // stats: 6 x 128 floats
  float *mean1 = st, *rstd1 = st + 128, *mean2 = st + 256, *rstd2 = st + 384,
        *mean3 = st + 512, *rstd3 = st + 640;

  // fc1 GEMM + BN + bf16 copy + row norms
  gemm_nt_wmma<<<256, 256, 0, stream>>>(x, fc1_w, fc1_b, bufA, MROWS, CH, CH, MROWS / 64);
  bn_stats<<<CH, 256, 0, stream>>>(bufA, mean1, rstd1);
  bn_apply_h<<<(MROWS * CH) / 256, 256, 0, stream>>>(bufA, mean1, rstd1, bn1_g, bn1_b, bufB, hbf);
  rowsq<<<MROWS / 8, 256, 0, stream>>>(bufB, xsq);

  // Fused KNN (bf16 WMMA distance GEMM + streaming top-9)
  knn_kernel<<<BATCH * (NPTS / 16), 128, 0, stream>>>(hbf, xsq, eidx);

  // Max-relative conv + BN + GELU
  maxrel<<<MROWS, 128, 0, stream>>>(bufB, eidx, cat);
  gemm_nt_wmma<<<256, 256, 0, stream>>>(cat, conv_w, conv_b, bufA, MROWS, CH, 2 * CH, MROWS / 64);
  bn_stats<<<CH, 256, 0, stream>>>(bufA, mean2, rstd2);
  bn_gelu<<<(MROWS * CH) / 256, 256, 0, stream>>>(bufA, mean2, rstd2, bnc_g, bnc_b, bufB);

  // fc2 + BN + residual
  gemm_nt_wmma<<<256, 256, 0, stream>>>(bufB, fc2_w, fc2_b, opre, MROWS, CH, CH, MROWS / 64);
  bn_stats<<<CH, 256, 0, stream>>>(opre, mean3, rstd3);
  bn_res<<<(MROWS * CH) / 256, 256, 0, stream>>>(opre, mean3, rstd3, bn2_g, bn2_b, x, (float*)d_out);
}
